// MultiHeadedAttention_25374666785120
// MI455X (gfx1250) — matrix operand order repaired and checked by reading, not compiled, not run on hardware
//
#include <hip/hip_runtime.h>
#include <hip/hip_bf16.h>

// ---------------------------------------------------------------------------
// MultiHeadedAttention for MI455X (gfx1250), bf16 WMMA everywhere.
//   B=2, S=2048, D=1024, H=16, DK=64
// Pipeline:
//   1) cvt fp32->bf16 for query/key/value and Wq/Wk/Wv/Wo
//   2) gemm_bf16_wmma (32x64 wave tiles, double-buffered):
//        Q = x@Wq^T+b -> [b,h,s,d];  K -> [b,h,s,d];  V -> [b,h,d,s] (transposed)
//   3) attn_wmma: 4 waves/block share one (b,h); K/V tiles double-buffered in
//      LDS via GLOBAL_LOAD_ASYNC_TO_LDS_B128 (ASYNCcnt) when available.
//   4) gemm_bf16_wmma mode 2: out = ctx@Wo^T + bo (fp32 to d_out)
// ---------------------------------------------------------------------------

typedef __attribute__((ext_vector_type(16))) __bf16 v16bf;
typedef __attribute__((ext_vector_type(8)))  __bf16 v8bf;
typedef __attribute__((ext_vector_type(8)))  float  v8f;

// Matches the builtin's declared parameter type on ROCm clang-22:
//   __attribute__((__vector_size__(4 * sizeof(int)))) int *
typedef int v4i_ls __attribute__((__vector_size__(4 * sizeof(int))));

constexpr int cH  = 16;
constexpr int cDK = 64;
constexpr int cS  = 2048;
constexpr int cB  = 2;
constexpr int cD  = cH * cDK;   // 1024
constexpr int cM  = cB * cS;    // 4096

union BF16x16 { v16bf v; v8bf h[2]; };

#if defined(__HIP_DEVICE_COMPILE__) && __has_builtin(__builtin_amdgcn_global_load_async_to_lds_b128)
#define USE_ASYNC_LDS 1
#else
#define USE_ASYNC_LDS 0
#endif

__device__ __forceinline__ __bf16 f2bf(float f) {
  unsigned u = __float_as_uint(f);
  u += 0x7FFFu + ((u >> 16) & 1u);          // round-to-nearest-even
  unsigned short hs = (unsigned short)(u >> 16);
  __bf16 r;
  __builtin_memcpy(&r, &hs, sizeof(r));
  return r;
}

// 16-byte global -> LDS copy; async (ASYNCcnt-tracked) on gfx1250 toolchains
// that expose the builtin, otherwise synchronous load+ds_store.
__device__ __forceinline__ void async_cp16(const __bf16* g, __bf16* l) {
#if USE_ASYNC_LDS
  __builtin_amdgcn_global_load_async_to_lds_b128((v4i_ls*)(g), (v4i_ls*)(l), 0, 0);
#else
  *(v8bf*)l = *(const v8bf*)g;
#endif
}

__device__ __forceinline__ void wait_async_copies() {
#if USE_ASYNC_LDS
#if __has_builtin(__builtin_amdgcn_s_wait_asynccnt)
  __builtin_amdgcn_s_wait_asynccnt(0);
#else
  asm volatile("s_wait_asynccnt 0" ::: "memory");
#endif
#endif
}

// ---------------------------------------------------------------------------
__global__ void cvt_f32_to_bf16(const float* __restrict__ in,
                                __bf16* __restrict__ out, int n) {
  int i = blockIdx.x * blockDim.x + threadIdx.x;
  int stride = gridDim.x * blockDim.x;
  for (; i < n; i += stride) out[i] = f2bf(in[i]);
}

// ---------------------------------------------------------------------------
// y = X[M,K] @ W[N,K]^T + bias[N].  Each wave: 32x64 tile, K ping-pong
// double-buffered so loads for the next 32-k slab overlap WMMAs on the
// current one (avoids s_wait_loadcnt 0 in the steady state).
// mode 0: bf16 out [((b*H+h)*S+s)*DK+d]; mode 1: bf16 out [((b*H+h)*DK+d)*S+s];
// mode 2: f32 out [r*N+c].
__global__ __launch_bounds__(128) void gemm_bf16_wmma(
    const __bf16* __restrict__ X, const __bf16* __restrict__ W,
    const float* __restrict__ bias, void* __restrict__ outp,
    int M, int N, int K, int mode) {
  const int lane = threadIdx.x & 31;
  const int wave = threadIdx.x >> 5;
  const int tile = blockIdx.x * 4 + wave;
  const int tn   = N >> 6;                 // 64-wide col tiles
  const int row0 = (tile / tn) << 5;       // 32-row tiles
  const int col0 = (tile % tn) << 6;
  if (row0 >= M) return;

  const int l16  = lane & 15;
  const int half = lane >> 4;
  const int aoff = half * 16;              // A-frag per-lane k offset
  const int boff = half * 16;              // B-frag per-lane k offset

  const __bf16* xrow0 = X + (size_t)(row0 + l16) * K;
  const __bf16* xrow1 = X + (size_t)(row0 + 16 + l16) * K;
  const __bf16* wrow[4];
#pragma unroll
  for (int nb = 0; nb < 4; ++nb)
    wrow[nb] = W + (size_t)(col0 + nb * 16 + l16) * K;

  v8f acc[2][4] = {};

  auto loadk = [&](int kb, BF16x16* a, BF16x16* b) {
    a[0].h[0] = *(const v8bf*)(xrow0 + kb + aoff);
    a[0].h[1] = *(const v8bf*)(xrow0 + kb + 8 + aoff);
    a[1].h[0] = *(const v8bf*)(xrow1 + kb + aoff);
    a[1].h[1] = *(const v8bf*)(xrow1 + kb + 8 + aoff);
#pragma unroll
    for (int nb = 0; nb < 4; ++nb) {
      b[nb].h[0] = *(const v8bf*)(wrow[nb] + kb + boff);
      b[nb].h[1] = *(const v8bf*)(wrow[nb] + kb + boff + 8);
    }
  };
  auto mm = [&](BF16x16* a, BF16x16* b) {
#pragma unroll
    for (int nb = 0; nb < 4; ++nb) {
      acc[0][nb] = __builtin_amdgcn_wmma_f32_16x16x32_bf16(
          false, a[0].v, false, b[nb].v, (short)0, acc[0][nb], false, false);
      acc[1][nb] = __builtin_amdgcn_wmma_f32_16x16x32_bf16(
          false, a[1].v, false, b[nb].v, (short)0, acc[1][nb], false, false);
    }
  };

  BF16x16 aP[2], bP[4], aQ[2], bQ[4];
  loadk(0, aP, bP);
  for (int kb = 0; kb < K; kb += 64) {
    loadk(kb + 32, aQ, bQ);                // prefetch slab kb+32
    mm(aP, bP);                            // compute slab kb
    if (kb + 64 < K) loadk(kb + 64, aP, bP);
    mm(aQ, bQ);
  }

  // C layout: VGPR j -> row row0 + i*16 + j + 8*half, col col0 + nb*16 + l16
#pragma unroll
  for (int i = 0; i < 2; ++i) {
#pragma unroll
    for (int nb = 0; nb < 4; ++nb) {
#pragma unroll
      for (int j = 0; j < 8; ++j) {
        const int r = row0 + i * 16 + j + half * 8;
        const int c = col0 + nb * 16 + l16;
        const float v = acc[i][nb][j] + bias[c];
        if (mode == 2) {
          ((float*)outp)[(size_t)r * N + c] = v;
        } else {
          const int bb = r >> 11, ss = r & (cS - 1);
          const int hh = c >> 6,  dd = c & (cDK - 1);
          const size_t idx = (mode == 0)
              ? (((size_t)(bb * cH + hh) * cS + ss) * cDK + dd)
              : (((size_t)(bb * cH + hh) * cDK + dd) * cS + ss);
          ((__bf16*)outp)[idx] = f2bf(v);
        }
      }
    }
  }
}

// ---------------------------------------------------------------------------
// Flash attention. One block = 4 waves, one (b,h), 64 query rows (16 per
// wave). K (32x64) and V^T (64x32) tiles are double-buffered in LDS and
// staged cooperatively with async global->LDS copies: copies for tile i+1 are
// in flight while all 4 waves compute tile i, drained by s_wait_asynccnt + a
// workgroup barrier.
__global__ __launch_bounds__(128) void attn_wmma(
    const __bf16* __restrict__ Qh,   // [B*H, S, DK]
    const __bf16* __restrict__ Kh,   // [B*H, S, DK]
    const __bf16* __restrict__ Vth,  // [B*H, DK, S]
    __bf16* __restrict__ ctx) {      // [B, S, H, DK] == row-major [M, D]
  __shared__ __bf16 Ktile[2][32 * 64];   // [sk][dk]
  __shared__ __bf16 Vtile[2][64 * 32];   // [dk][sk]
  __shared__ __bf16 plds[4][16 * 32];

  const int tid  = threadIdx.x;
  const int lane = tid & 31;
  const int wave = tid >> 5;
  const int bh = blockIdx.x >> 5;          // 32 q-blocks per (b,h)
  const int qb = blockIdx.x & 31;
  const int qrow0 = qb * 64 + wave * 16;
  const int l16  = lane & 15;
  const int half = lane >> 4;
  const int aoff = half * 16;
  const int boff = half * 16;

  const __bf16* Kp = Kh + (size_t)bh * cS * cDK;
  const __bf16* Vp = Vth + (size_t)bh * cDK * cS;

  // cooperative stage of one 32-key tile (K: 4KB contiguous, V^T: 64 rows x 64B)
  auto stage = [&](int sk0, __bf16* kbuf, __bf16* vbuf) {
    const __bf16* kg = Kp + (size_t)sk0 * cDK;
#pragma unroll
    for (int c = tid; c < 256; c += 128)
      async_cp16(kg + c * 8, kbuf + c * 8);
#pragma unroll
    for (int c = tid; c < 256; c += 128) {
      const int row = c >> 2, off = (c & 3) * 8;
      async_cp16(Vp + (size_t)row * cS + sk0 + off, vbuf + row * 32 + off);
    }
  };

  // Q fragments (dk 0..31 and 32..63), loaded once
  BF16x16 q0, q1;
  {
    const __bf16* qr = Qh + ((size_t)bh * cS + qrow0 + l16) * cDK;
    q0.h[0] = *(const v8bf*)(qr + aoff);
    q0.h[1] = *(const v8bf*)(qr + 8 + aoff);
    q1.h[0] = *(const v8bf*)(qr + 32 + aoff);
    q1.h[1] = *(const v8bf*)(qr + 40 + aoff);
  }

  float m[8], l[8];
  v8f o[4] = {};
#pragma unroll
  for (int j = 0; j < 8; ++j) { m[j] = -1e30f; l[j] = 0.0f; }

  stage(0, Ktile[0], Vtile[0]);
  wait_async_copies();
  __syncthreads();

  for (int it = 0; it < cS / 32; ++it) {
    const int cur = it & 1;
    if (it + 1 < cS / 32)
      stage((it + 1) * 32, Ktile[cur ^ 1], Vtile[cur ^ 1]);

    const __bf16* Kt = &Ktile[cur][0];
    const __bf16* Vt = &Vtile[cur][0];

    // ---- scores: two 16x16 tiles (key cols 0..15 and 16..31 of this slab)
    v8f s0 = {}, s1 = {};
    {
      const __bf16* kr0 = Kt + (size_t)l16 * 64;
      const __bf16* kr1 = Kt + (size_t)(16 + l16) * 64;
      BF16x16 b;
      b.h[0] = *(const v8bf*)(kr0 + boff);
      b.h[1] = *(const v8bf*)(kr0 + boff + 8);
      s0 = __builtin_amdgcn_wmma_f32_16x16x32_bf16(false, q0.v, false, b.v, (short)0, s0, false, false);
      b.h[0] = *(const v8bf*)(kr0 + 32 + boff);
      b.h[1] = *(const v8bf*)(kr0 + 32 + boff + 8);
      s0 = __builtin_amdgcn_wmma_f32_16x16x32_bf16(false, q1.v, false, b.v, (short)0, s0, false, false);
      b.h[0] = *(const v8bf*)(kr1 + boff);
      b.h[1] = *(const v8bf*)(kr1 + boff + 8);
      s1 = __builtin_amdgcn_wmma_f32_16x16x32_bf16(false, q0.v, false, b.v, (short)0, s1, false, false);
      b.h[0] = *(const v8bf*)(kr1 + 32 + boff);
      b.h[1] = *(const v8bf*)(kr1 + 32 + boff + 8);
      s1 = __builtin_amdgcn_wmma_f32_16x16x32_bf16(false, q1.v, false, b.v, (short)0, s1, false, false);
    }

    // ---- online softmax; element j = row j + 8*half, columns across 16 lanes
    float alpha[8];
#pragma unroll
    for (int j = 0; j < 8; ++j) {
      float a0 = s0[j] * 0.125f;            // 1/sqrt(DK)
      float a1 = s1[j] * 0.125f;
      float mt = fmaxf(a0, a1);
      mt = fmaxf(mt, __shfl_xor(mt, 1));
      mt = fmaxf(mt, __shfl_xor(mt, 2));
      mt = fmaxf(mt, __shfl_xor(mt, 4));
      mt = fmaxf(mt, __shfl_xor(mt, 8));
      const float mn = fmaxf(m[j], mt);
      alpha[j] = __expf(m[j] - mn);
      const float p0 = __expf(a0 - mn);
      const float p1 = __expf(a1 - mn);
      s0[j] = p0; s1[j] = p1;
      float rs = p0 + p1;
      rs += __shfl_xor(rs, 1);
      rs += __shfl_xor(rs, 2);
      rs += __shfl_xor(rs, 4);
      rs += __shfl_xor(rs, 8);
      l[j] = l[j] * alpha[j] + rs;
      m[j] = mn;
    }
#pragma unroll
    for (int nb = 0; nb < 4; ++nb)
#pragma unroll
      for (int j = 0; j < 8; ++j) o[nb][j] *= alpha[j];

    // ---- P tile C-layout -> A-layout via per-wave LDS (16x32 bf16)
    __bf16* pw = &plds[wave][0];
#pragma unroll
    for (int j = 0; j < 8; ++j) {
      const int r = j + half * 8;
      pw[r * 32 + l16]      = f2bf(s0[j]);
      pw[r * 32 + 16 + l16] = f2bf(s1[j]);
    }
    asm volatile("s_wait_dscnt 0" ::: "memory");   // same-wave DS ordering
    BF16x16 pa;
    {
      const __bf16* pr = pw + l16 * 32 + aoff;
      pa.h[0] = *(const v8bf*)(pr);
      pa.h[1] = *(const v8bf*)(pr + 8);
    }

    // ---- O += P @ V : B frags contiguous from V^T tile rows (d-major)
#pragma unroll
    for (int nb = 0; nb < 4; ++nb) {
      const __bf16* vr = Vt + (size_t)(nb * 16 + l16) * 32 + boff;
      BF16x16 b;
      b.h[0] = *(const v8bf*)(vr);
      b.h[1] = *(const v8bf*)(vr + 8);
      o[nb] = __builtin_amdgcn_wmma_f32_16x16x32_bf16(
          false, pa.v, false, b.v, (short)0, o[nb], false, false);
    }

    wait_async_copies();    // next tile's copies finished during this compute
    __syncthreads();
  }

  // ---- finalize: ctx[b, s, h, d] = O / l
  const int bb = bh >> 4, hh = bh & 15;
#pragma unroll
  for (int j = 0; j < 8; ++j) {
    const float inv = 1.0f / l[j];
    const int qrow = qrow0 + j + half * 8;
#pragma unroll
    for (int nb = 0; nb < 4; ++nb) {
      const int d = nb * 16 + l16;
      ctx[((size_t)(bb * cS + qrow) * cH + hh) * cDK + d] = f2bf(o[nb][j] * inv);
    }
  }
}

// ---------------------------------------------------------------------------
extern "C" void kernel_launch(void* const* d_in, const int* in_sizes, int n_in,
                              void* d_out, int out_size, void* d_ws, size_t ws_size,
                              hipStream_t stream) {
  (void)in_sizes; (void)n_in; (void)out_size; (void)ws_size;
  const float* query = (const float*)d_in[0];
  const float* key   = (const float*)d_in[1];
  const float* value = (const float*)d_in[2];
  const float* Wq = (const float*)d_in[3];
  const float* bq = (const float*)d_in[4];
  const float* Wk = (const float*)d_in[5];
  const float* bk = (const float*)d_in[6];
  const float* Wv = (const float*)d_in[7];
  const float* bv = (const float*)d_in[8];
  const float* Wo = (const float*)d_in[9];
  const float* bo = (const float*)d_in[10];
  float* out = (float*)d_out;

  char* p = (char*)d_ws;
  auto take = [&](size_t bytes) {
    char* r = p; p += (bytes + 255) & ~(size_t)255; return r;
  };
  const size_t actB = (size_t)cM * cD * sizeof(__bf16);  // 8 MiB
  const size_t wB   = (size_t)cD * cD * sizeof(__bf16);  // 2 MiB
  __bf16* Xq  = (__bf16*)take(actB);
  __bf16* Xk  = (__bf16*)take(actB);
  __bf16* Xv  = (__bf16*)take(actB);
  __bf16* Wqb = (__bf16*)take(wB);
  __bf16* Wkb = (__bf16*)take(wB);
  __bf16* Wvb = (__bf16*)take(wB);
  __bf16* Wob = (__bf16*)take(wB);
  __bf16* Qh  = (__bf16*)take(actB);
  __bf16* Kh  = (__bf16*)take(actB);
  __bf16* Vt  = (__bf16*)take(actB);
  __bf16* ctx = (__bf16*)take(actB);

  const int nAct = cM * cD;   // 4,194,304
  const int nW   = cD * cD;   // 1,048,576
  cvt_f32_to_bf16<<<4096, 256, 0, stream>>>(query, Xq, nAct);
  cvt_f32_to_bf16<<<4096, 256, 0, stream>>>(key,   Xk, nAct);
  cvt_f32_to_bf16<<<4096, 256, 0, stream>>>(value, Xv, nAct);
  cvt_f32_to_bf16<<<1024, 256, 0, stream>>>(Wq, Wqb, nW);
  cvt_f32_to_bf16<<<1024, 256, 0, stream>>>(Wk, Wkb, nW);
  cvt_f32_to_bf16<<<1024, 256, 0, stream>>>(Wv, Wvb, nW);
  cvt_f32_to_bf16<<<1024, 256, 0, stream>>>(Wo, Wob, nW);

  // (4096/32)*(1024/64) = 2048 32x64 tiles, 4 waves per block -> 512 blocks
  const int gemmBlocks = (cM / 32) * (cD / 64) / 4;
  gemm_bf16_wmma<<<gemmBlocks, 128, 0, stream>>>(Xq, Wqb, bq, Qh, cM, cD, cD, 0);
  gemm_bf16_wmma<<<gemmBlocks, 128, 0, stream>>>(Xk, Wkb, bk, Kh, cM, cD, cD, 0);
  gemm_bf16_wmma<<<gemmBlocks, 128, 0, stream>>>(Xv, Wvb, bv, Vt, cM, cD, cD, 1);

  // B*H * (S/64 q-rows per block) = 32 * 32 = 1024 blocks of 4 waves
  attn_wmma<<<cB * cH * (cS / 64), 128, 0, stream>>>(Qh, Kh, Vt, ctx);

  gemm_bf16_wmma<<<gemmBlocks, 128, 0, stream>>>(ctx, Wob, bo, out, cM, cD, cD, 2);
}
